// GlobalTransformerPretrain_7842610282634
// MI455X (gfx1250) — matrix operand order repaired and checked by reading, not compiled, not run on hardware
//
#include <hip/hip_runtime.h>
#include <hip/hip_bf16.h>
#include <math.h>

typedef __bf16 bf16;
typedef bf16 v8bf  __attribute__((ext_vector_type(8)));
typedef bf16 v16bf __attribute__((ext_vector_type(16)));
typedef float v8f  __attribute__((ext_vector_type(8)));

#if defined(__has_builtin)
#  if __has_builtin(__builtin_amdgcn_tensor_load_to_lds) && __has_builtin(__builtin_amdgcn_s_wait_tensorcnt)
#    define HAVE_TDM 1
#  endif
#endif
#ifndef HAVE_TDM
#  define HAVE_TDM 0
#endif

#define NN 8192
#define DD 512
#define FDIM 2048
#define MM 4096
#define KIDX 128

#define BM 128
#define BN 256
#define BK 32
#define LDT 40   // padded LDS row stride (bf16): 80B rows (64B data + 16B pad)

#if HAVE_TDM
typedef unsigned tdm_v4u __attribute__((ext_vector_type(4)));
typedef int      tdm_v8i __attribute__((ext_vector_type(8)));
typedef int      tdm_v4i __attribute__((ext_vector_type(4)));

// DMA `rows` rows of 32 bf16 (64B) from global (row stride = stride_elems*2B) into LDS
// at lds_addr, inserting 16B padding after every 64B row (matches LDT=40 layout).
// D# fields per CDNA5 ISA.  6-arg builtin: groups 0..3 + VADDR4
// slot (unused -> zeros) + cache policy.
__device__ __forceinline__ void tdm_load_rows(const bf16* gptr, unsigned lds_addr,
                                              unsigned rows, unsigned stride_elems) {
  unsigned long long ga = (unsigned long long)(uintptr_t)gptr;
  tdm_v4u g0;
  g0[0] = 1u;                                           // count=1, user mode, no gather
  g0[1] = lds_addr;                                     // lds_addr [63:32]
  g0[2] = (unsigned)(ga & 0xFFFFFFFFu);                 // global_addr low
  g0[3] = (unsigned)((ga >> 32) & 0x01FFFFFFu) | (2u << 30);  // global_addr hi, type=2
  tdm_v8i g1;
  // [17:16] data_size=1 (2B), [20] pad_enable, [24:22] pad_interval=3 (16 dwords),
  // [31:25] pad_amount=3 (4 dwords); workgroup_mask=0
  unsigned w0 = (1u << 16) | (1u << 20) | (3u << 22) | (3u << 25);
  unsigned t0 = stride_elems;        // tensor_dim0 (row length avail) = full row stride
  unsigned t1 = rows;                // tensor_dim1
  g1[0] = (int)w0;
  g1[1] = (int)((t0 & 0xFFFFu) << 16);                  // abar_addr=0 | tensor_dim0 lo16
  g1[2] = (int)((t0 >> 16) | ((t1 & 0xFFFFu) << 16));   // tensor_dim0 hi16 | tensor_dim1 lo16
  g1[3] = (int)((t1 >> 16) | (32u << 16));              // tensor_dim1 hi16 | tile_dim0=32
  g1[4] = (int)(rows & 0xFFFFu);                        // tile_dim1=rows, tile_dim2=0
  g1[5] = (int)stride_elems;                            // tensor_dim0_stride lo32
  g1[6] = 0;                                            // stride hi16 | dim1_stride lo16
  g1[7] = 0;
  tdm_v4i z4 = {0, 0, 0, 0};
  tdm_v8i z8 = {0, 0, 0, 0, 0, 0, 0, 0};
  __builtin_amdgcn_tensor_load_to_lds(g0, g1, z4, z4, z8, 0);
}
#endif

// ---------------- elementwise / prep ----------------
__global__ void prep_kernel(const float* __restrict__ x, float* __restrict__ uxf,
                            bf16* __restrict__ uxb, int n) {
  int i = blockIdx.x * blockDim.x + threadIdx.x;
  if (i < n) { float v = x[i]; uxf[i] = v; uxb[i] = (bf16)v; }
}

__global__ void f2b_kernel(const float* __restrict__ s, bf16* __restrict__ d, int n) {
  int i = blockIdx.x * blockDim.x + threadIdx.x;
  if (i < n) d[i] = (bf16)s[i];
}

__global__ void mask_gather_kernel(const float* __restrict__ x, const int* __restrict__ rows,
                                   const int* __restrict__ cols, float* __restrict__ uxf,
                                   bf16* __restrict__ uxb, float* __restrict__ x_init) {
  int i = blockIdx.x * blockDim.x + threadIdx.x;   // i = m*KIDX + k
  int m = i / KIDX;
  int r = rows[m];
  int c = cols[i];
  size_t o = (size_t)r * DD + (size_t)c;
  x_init[i] = x[o];
  uxf[o] = 0.0f;
  uxb[o] = (bf16)0.0f;
}

__global__ void gather_out_kernel(const float* __restrict__ h, const int* __restrict__ rows,
                                  const int* __restrict__ cols, float* __restrict__ x_rec) {
  int i = blockIdx.x * blockDim.x + threadIdx.x;
  int m = i / KIDX;
  int r = rows[m];
  int c = cols[i];
  x_rec[i] = h[(size_t)r * DD + (size_t)c];
}

// ---------------- row softmax (fp32 in-place + bf16 mirror) ----------------
__global__ __launch_bounds__(256) void softmax_f32_kernel(float* __restrict__ S,
                                                          bf16* __restrict__ P) {
  __shared__ float red[256];
  const int row = blockIdx.x, tid = threadIdx.x;
  float* sp = S + (size_t)row * NN;
  bf16*  pp = P + (size_t)row * NN;
  float loc[NN / 256];
  float mx = -3.0e38f;
#pragma unroll
  for (int i = 0; i < NN / 256; ++i) { float v = sp[tid + (i << 8)]; loc[i] = v; mx = fmaxf(mx, v); }
  red[tid] = mx; __syncthreads();
  for (int o = 128; o > 0; o >>= 1) { if (tid < o) red[tid] = fmaxf(red[tid], red[tid + o]); __syncthreads(); }
  mx = red[0]; __syncthreads();
  float sum = 0.0f;
#pragma unroll
  for (int i = 0; i < NN / 256; ++i) { float e = expf(loc[i] - mx); loc[i] = e; sum += e; }
  red[tid] = sum; __syncthreads();
  for (int o = 128; o > 0; o >>= 1) { if (tid < o) red[tid] += red[tid + o]; __syncthreads(); }
  float inv = 1.0f / red[0];
#pragma unroll
  for (int i = 0; i < NN / 256; ++i) {
    float v = loc[i] * inv;
    sp[tid + (i << 8)] = v;
    pp[tid + (i << 8)] = (bf16)v;
  }
}

// bf16 scores -> bf16 probs, in place
__global__ __launch_bounds__(256) void softmax_b16_kernel(bf16* __restrict__ S) {
  __shared__ float red[256];
  const int row = blockIdx.x, tid = threadIdx.x;
  bf16* sp = S + (size_t)row * NN;
  float loc[NN / 256];
  float mx = -3.0e38f;
#pragma unroll
  for (int i = 0; i < NN / 256; ++i) { float v = (float)sp[tid + (i << 8)]; loc[i] = v; mx = fmaxf(mx, v); }
  red[tid] = mx; __syncthreads();
  for (int o = 128; o > 0; o >>= 1) { if (tid < o) red[tid] = fmaxf(red[tid], red[tid + o]); __syncthreads(); }
  mx = red[0]; __syncthreads();
  float sum = 0.0f;
#pragma unroll
  for (int i = 0; i < NN / 256; ++i) { float e = expf(loc[i] - mx); loc[i] = e; sum += e; }
  red[tid] = sum; __syncthreads();
  for (int o = 128; o > 0; o >>= 1) { if (tid < o) red[tid] += red[tid + o]; __syncthreads(); }
  float inv = 1.0f / red[0];
#pragma unroll
  for (int i = 0; i < NN / 256; ++i) sp[tid + (i << 8)] = (bf16)(loc[i] * inv);
}

// ---------------- residual add + LayerNorm (D=512, 256 thr -> 2 elems/thr) ----------------
__global__ __launch_bounds__(256) void add_ln_kernel(const float* __restrict__ a,
                                                     const float* __restrict__ b,
                                                     const float* __restrict__ g,
                                                     const float* __restrict__ be,
                                                     float* __restrict__ outF,
                                                     bf16* __restrict__ outB) {
  __shared__ float red[256];
  const int row = blockIdx.x, tid = threadIdx.x;
  const float* pa = a + (size_t)row * DD;
  const float* pb = b + (size_t)row * DD;
  float h0 = pa[tid] + pb[tid];
  float h1 = pa[tid + 256] + pb[tid + 256];
  red[tid] = h0 + h1; __syncthreads();
  for (int o = 128; o > 0; o >>= 1) { if (tid < o) red[tid] += red[tid + o]; __syncthreads(); }
  float mu = red[0] * (1.0f / DD); __syncthreads();
  float d0 = h0 - mu, d1 = h1 - mu;
  red[tid] = d0 * d0 + d1 * d1; __syncthreads();
  for (int o = 128; o > 0; o >>= 1) { if (tid < o) red[tid] += red[tid + o]; __syncthreads(); }
  float rs = rsqrtf(red[0] * (1.0f / DD) + 1e-5f);
  float v0 = d0 * rs * g[tid] + be[tid];
  float v1 = d1 * rs * g[tid + 256] + be[tid + 256];
  outF[(size_t)row * DD + tid] = v0;
  outF[(size_t)row * DD + tid + 256] = v1;
  if (outB) {
    outB[(size_t)row * DD + tid] = (bf16)v0;
    outB[(size_t)row * DD + tid + 256] = (bf16)v1;
  }
}

// ---------------- bf16 WMMA GEMM (double-buffered; TDM-staged where possible) ----------------
// C[128 x 256 tile] = act(alpha * A @ B(^T) + bias)
// BT=true:  B is [Ncols x Kdim] row-major (compute A @ B^T); both tiles are
//           row-gathers of 32 contiguous bf16 -> staged by the Tensor Data Mover.
// BT=false: B is [Kdim x Ncols] row-major; B tile needs a transpose -> manual staging,
//           A tile still TDM.
// 8 waves in a 2x4 grid of 64x64 wave tiles; each wave: 4x4 wmma accumulators.
template <bool BT>
__global__ __launch_bounds__(256) void gemm_bf16_kernel(const bf16* __restrict__ A,
                                                        const bf16* __restrict__ B,
                                                        const float* __restrict__ bias,
                                                        float* __restrict__ outF,
                                                        bf16* __restrict__ outB,
                                                        int Kdim, int Ncols,
                                                        float alpha, int flags) {
  __shared__ bf16 aT[2][BM * LDT];
  __shared__ bf16 bT[2][BN * LDT];

  const int tid = threadIdx.x;
  const int lane = tid & 31;
  const int w  = tid >> 5;          // 0..7
  const int wr = w >> 2;            // 0..1  (row group of 64)
  const int wc = w & 3;             // 0..3  (col group of 64)
  const int lh = lane >> 4;         // lane half
  const int lm = lane & 15;

  const size_t row0 = (size_t)blockIdx.y * BM;
  const size_t col0 = (size_t)blockIdx.x * BN;

  v8f acc[4][4];
#pragma unroll
  for (int r = 0; r < 4; ++r)
#pragma unroll
    for (int c = 0; c < 4; ++c) acc[r][c] = (v8f)0.0f;

  // staging coordinates (manual paths)
  const int arow = tid >> 1;
  const int akk  = (tid & 1) << 4;
  const int bkk2 = (tid & 15) << 1;
  const int bnb  = (tid >> 4) << 4;

  v8bf ar0, ar1;                 // A staging regs (manual fallback)
  v8bf br0, br1, br2, br3;       // B staging regs

#if HAVE_TDM
  const unsigned a_lds0 = (unsigned)(uintptr_t)&aT[0][0];
  const unsigned a_lds1 = (unsigned)(uintptr_t)&aT[1][0];
  const unsigned b_lds0 = (unsigned)(uintptr_t)&bT[0][0];
  const unsigned b_lds1 = (unsigned)(uintptr_t)&bT[1][0];
#endif

  auto load_tile = [&](int k0, int buf) {
#if HAVE_TDM
    if (w == 0)
      tdm_load_rows(A + row0 * (size_t)Kdim + (size_t)k0,
                    buf ? a_lds1 : a_lds0, BM, (unsigned)Kdim);
    if (BT) {
      if (w == 1)
        tdm_load_rows(B + col0 * (size_t)Kdim + (size_t)k0,
                      buf ? b_lds1 : b_lds0, BN, (unsigned)Kdim);
    } else {
      const bf16* s0 = B + (size_t)(k0 + bkk2) * (size_t)Ncols + col0 + (size_t)bnb;
      const bf16* s1 = s0 + (size_t)Ncols;
      br0 = *(const v8bf*)s0;
      br1 = *(const v8bf*)(s0 + 8);
      br2 = *(const v8bf*)s1;
      br3 = *(const v8bf*)(s1 + 8);
    }
#else
    (void)buf;
    {
      const bf16* src = A + (row0 + (size_t)arow) * (size_t)Kdim + (size_t)(k0 + akk);
      ar0 = *(const v8bf*)src;
      ar1 = *(const v8bf*)(src + 8);
    }
    if (BT) {
      const bf16* src = B + (col0 + (size_t)tid) * (size_t)Kdim + (size_t)k0;
      br0 = *(const v8bf*)src;
      br1 = *(const v8bf*)(src + 8);
      br2 = *(const v8bf*)(src + 16);
      br3 = *(const v8bf*)(src + 24);
    } else {
      const bf16* s0 = B + (size_t)(k0 + bkk2) * (size_t)Ncols + col0 + (size_t)bnb;
      const bf16* s1 = s0 + (size_t)Ncols;
      br0 = *(const v8bf*)s0;
      br1 = *(const v8bf*)(s0 + 8);
      br2 = *(const v8bf*)s1;
      br3 = *(const v8bf*)(s1 + 8);
    }
#endif
  };

  auto store_tile = [&](int buf) {
#if !HAVE_TDM
    {
      bf16* dst = &aT[buf][arow * LDT + akk];
      *(v8bf*)dst       = ar0;
      *(v8bf*)(dst + 8) = ar1;
    }
    if (BT) {
      bf16* dst = &bT[buf][tid * LDT];
      *(v8bf*)dst        = br0;
      *(v8bf*)(dst + 8)  = br1;
      *(v8bf*)(dst + 16) = br2;
      *(v8bf*)(dst + 24) = br3;
    }
#endif
    if (!BT) {
      // pack (k, k+1) pairs -> one dword per n; LDS layout is n-major so pairs are contiguous
#pragma unroll
      for (int i = 0; i < 16; ++i) {
        bf16 lo = (i < 8) ? br0[i] : br1[i - 8];
        bf16 hi = (i < 8) ? br2[i] : br3[i - 8];
        union { bf16 h[2]; unsigned u; } p;
        p.h[0] = lo; p.h[1] = hi;
        *(unsigned*)&bT[buf][(bnb + i) * LDT + bkk2] = p.u;
      }
    }
#if HAVE_TDM
    if (w == 0 || (BT && w == 1)) __builtin_amdgcn_s_wait_tensorcnt(0);
#endif
  };

  load_tile(0, 0);
  store_tile(0);
  __syncthreads();

  const int nk = Kdim / BK;
  for (int t = 0; t < nk; ++t) {
    const int cur = t & 1;
    if (t + 1 < nk) load_tile((t + 1) * BK, cur ^ 1);   // async: TDM DMA + reg loads

    // ---- fragments per ISA wave32 layouts ----
    v16bf af[4];
#pragma unroll
    for (int r = 0; r < 4; ++r) {
      const bf16* p = &aT[cur][(wr * 64 + r * 16 + lm) * LDT + lh * 8];
      v8bf lo = *(const v8bf*)p;            // K = lh*8 .. +7
      v8bf hi = *(const v8bf*)(p + 16);     // K = 16+lh*8 .. +7
#pragma unroll
      for (int i = 0; i < 8; ++i) { af[r][i] = lo[i]; af[r][8 + i] = hi[i]; }
    }
    v16bf bfm[4];
#pragma unroll
    for (int c = 0; c < 4; ++c) {
      const bf16* p = &bT[cur][(wc * 64 + c * 16 + lm) * LDT + lh * 8];
      v8bf lo = *(const v8bf*)p;            // K = lh*8 .. +7
      v8bf hi = *(const v8bf*)(p + 16);     // K = 16+lh*8 .. +7
#pragma unroll
      for (int i = 0; i < 8; ++i) { bfm[c][i] = lo[i]; bfm[c][8 + i] = hi[i]; }
    }
#pragma unroll
    for (int r = 0; r < 4; ++r)
#pragma unroll
      for (int c = 0; c < 4; ++c)
        acc[r][c] = __builtin_amdgcn_wmma_f32_16x16x32_bf16(
            false, af[r], false, bfm[c], (short)0, acc[r][c], false, false);

    if (t + 1 < nk) store_tile(cur ^ 1);   // drain regs / wait TENSORcnt for next buffer
    __syncthreads();
  }

  // ---- epilogue: C/D layout (VGPR j -> row lh*8+j, col lm) ----
  const bool do_gelu = (flags & 2) != 0;
#pragma unroll
  for (int r = 0; r < 4; ++r)
#pragma unroll
    for (int c = 0; c < 4; ++c)
#pragma unroll
      for (int j = 0; j < 8; ++j) {
        size_t grow = row0 + (size_t)(wr * 64 + r * 16 + lh * 8 + j);
        size_t gcol = col0 + (size_t)(wc * 64 + c * 16 + lm);
        float v = acc[r][c][j] * alpha;
        if (bias) v += bias[gcol];
        if (do_gelu) v = 0.5f * v * (1.0f + erff(v * 0.70710678118654752f));
        size_t o = grow * (size_t)Ncols + gcol;
        if (outF) outF[o] = v;
        if (outB) outB[o] = (bf16)v;
      }
}

// ---------------- host launch ----------------
extern "C" void kernel_launch(void* const* d_in, const int* in_sizes, int n_in,
                              void* d_out, int out_size, void* d_ws, size_t ws_size,
                              hipStream_t stream) {
  (void)in_sizes; (void)n_in; (void)out_size; (void)ws_size;
  const float* x        = (const float*)d_in[0];
  const int*   mask_nd  = (const int*)d_in[1];
  const int*   gene_idx = (const int*)d_in[2];
  const float* Wq_e = (const float*)d_in[3];
  const float* Wk_e = (const float*)d_in[4];
  const float* Wv_e = (const float*)d_in[5];
  const float* ln1_g = (const float*)d_in[6];
  const float* ln1_b = (const float*)d_in[7];
  const float* ff_W1 = (const float*)d_in[8];
  const float* ff_b1 = (const float*)d_in[9];
  const float* ff_W2 = (const float*)d_in[10];
  const float* ff_b2 = (const float*)d_in[11];
  const float* ln2_g = (const float*)d_in[12];
  const float* ln2_b = (const float*)d_in[13];
  const float* Wq_d = (const float*)d_in[14];
  const float* Wk_d = (const float*)d_in[15];
  const float* Wv_d = (const float*)d_in[16];
  const float* ln3_g = (const float*)d_in[17];
  const float* ln3_b = (const float*)d_in[18];
  const float* ln4_g = (const float*)d_in[19];
  const float* ln4_b = (const float*)d_in[20];
  const float* head_W = (const float*)d_in[21];
  const float* head_b = (const float*)d_in[22];

  float* out = (float*)d_out;
  float* o_xinit = out;
  float* o_xrec  = out + (size_t)MM * KIDX;
  float* o_enc   = out + (size_t)2 * MM * KIDX;
  float* o_recon = o_enc + (size_t)NN * NN;

  // workspace carve-up
  char* base = (char*)d_ws;
  size_t off = 0;
  auto carve = [&](size_t bytes) -> char* {
    char* p = base + off;
    off += (bytes + 255) & ~(size_t)255;
    return p;
  };
  float* use_x_f = (float*)carve((size_t)NN * DD * 4);
  bf16*  uxb     = (bf16*) carve((size_t)NN * DD * 2);
  bf16*  Qb      = (bf16*) carve((size_t)NN * DD * 2);
  bf16*  Kb      = (bf16*) carve((size_t)NN * DD * 2);
  bf16*  Vb      = (bf16*) carve((size_t)NN * DD * 2);
  bf16*  Pb      = (bf16*) carve((size_t)NN * NN * 2);
  float* tA      = (float*)carve((size_t)NN * DD * 4);
  float* tB      = (float*)carve((size_t)NN * DD * 4);
  float* tC      = (float*)carve((size_t)NN * DD * 4);
  bf16*  hb      = (bf16*) carve((size_t)NN * DD * 2);
  bf16*  t1b     = (bf16*) carve((size_t)NN * FDIM * 2);
  bf16*  wqeb    = (bf16*) carve((size_t)DD * DD * 2);
  bf16*  wkeb    = (bf16*) carve((size_t)DD * DD * 2);
  bf16*  wveb    = (bf16*) carve((size_t)DD * DD * 2);
  bf16*  w1b     = (bf16*) carve((size_t)DD * FDIM * 2);
  bf16*  w2b     = (bf16*) carve((size_t)FDIM * DD * 2);
  bf16*  wqdb    = (bf16*) carve((size_t)DD * DD * 2);
  bf16*  wkdb    = (bf16*) carve((size_t)DD * DD * 2);
  bf16*  wvdb    = (bf16*) carve((size_t)DD * DD * 2);
  bf16*  hwb     = (bf16*) carve((size_t)DD * DD * 2);

  const int T = 256;
  const float ALPHA_S = 0.04419417382415922f;  // 1/sqrt(512)
  dim3 g_qkv(DD / BN, NN / BM);     // (2, 64)
  dim3 g_s(NN / BN, NN / BM);       // (32, 64)
  dim3 g_ff1(FDIM / BN, NN / BM);   // (8, 64)

  // prep + mask
  prep_kernel<<<(NN * DD) / T, T, 0, stream>>>(x, use_x_f, uxb, NN * DD);
  mask_gather_kernel<<<(MM * KIDX) / T, T, 0, stream>>>(x, mask_nd, gene_idx, use_x_f, uxb, o_xinit);

  // weights -> bf16
  f2b_kernel<<<(DD * DD) / T, T, 0, stream>>>(Wq_e, wqeb, DD * DD);
  f2b_kernel<<<(DD * DD) / T, T, 0, stream>>>(Wk_e, wkeb, DD * DD);
  f2b_kernel<<<(DD * DD) / T, T, 0, stream>>>(Wv_e, wveb, DD * DD);
  f2b_kernel<<<(DD * FDIM) / T, T, 0, stream>>>(ff_W1, w1b, DD * FDIM);
  f2b_kernel<<<(FDIM * DD) / T, T, 0, stream>>>(ff_W2, w2b, FDIM * DD);
  f2b_kernel<<<(DD * DD) / T, T, 0, stream>>>(Wq_d, wqdb, DD * DD);
  f2b_kernel<<<(DD * DD) / T, T, 0, stream>>>(Wk_d, wkdb, DD * DD);
  f2b_kernel<<<(DD * DD) / T, T, 0, stream>>>(Wv_d, wvdb, DD * DD);
  f2b_kernel<<<(DD * DD) / T, T, 0, stream>>>(head_W, hwb, DD * DD);

  // ---- encoder attention ----
  gemm_bf16_kernel<false><<<g_qkv, T, 0, stream>>>(uxb, wqeb, nullptr, nullptr, Qb, DD, DD, 1.0f, 0);
  gemm_bf16_kernel<false><<<g_qkv, T, 0, stream>>>(uxb, wkeb, nullptr, nullptr, Kb, DD, DD, 1.0f, 0);
  gemm_bf16_kernel<false><<<g_qkv, T, 0, stream>>>(uxb, wveb, nullptr, nullptr, Vb, DD, DD, 1.0f, 0);
  gemm_bf16_kernel<true><<<g_s, T, 0, stream>>>(Qb, Kb, nullptr, o_enc, nullptr, DD, NN, ALPHA_S, 0);
  softmax_f32_kernel<<<NN, T, 0, stream>>>(o_enc, Pb);
  gemm_bf16_kernel<false><<<g_qkv, T, 0, stream>>>(Pb, Vb, nullptr, tA, nullptr, NN, DD, 1.0f, 0);
  add_ln_kernel<<<NN, T, 0, stream>>>(tA, use_x_f, ln1_g, ln1_b, tB, hb);

  // ---- encoder FFN ----
  gemm_bf16_kernel<false><<<g_ff1, T, 0, stream>>>(hb, w1b, ff_b1, nullptr, t1b, DD, FDIM, 1.0f, 2);
  gemm_bf16_kernel<false><<<g_qkv, T, 0, stream>>>(t1b, w2b, ff_b2, tA, nullptr, FDIM, DD, 1.0f, 0);
  add_ln_kernel<<<NN, T, 0, stream>>>(tB, tA, ln2_g, ln2_b, tC, hb);

  // ---- decoder attention ----
  gemm_bf16_kernel<false><<<g_qkv, T, 0, stream>>>(hb, wqdb, nullptr, nullptr, Qb, DD, DD, 1.0f, 0);
  gemm_bf16_kernel<false><<<g_qkv, T, 0, stream>>>(hb, wkdb, nullptr, nullptr, Kb, DD, DD, 1.0f, 0);
  gemm_bf16_kernel<false><<<g_qkv, T, 0, stream>>>(hb, wvdb, nullptr, nullptr, Vb, DD, DD, 1.0f, 0);
  gemm_bf16_kernel<true><<<g_s, T, 0, stream>>>(Qb, Kb, nullptr, nullptr, Pb, DD, NN, ALPHA_S, 0);
  softmax_b16_kernel<<<NN, T, 0, stream>>>(Pb);
  gemm_bf16_kernel<false><<<g_qkv, T, 0, stream>>>(Pb, Vb, nullptr, tA, nullptr, NN, DD, 1.0f, 0);
  add_ln_kernel<<<NN, T, 0, stream>>>(tA, tC, ln3_g, ln3_b, tB, hb);

  // ---- decoder FFN ----
  gemm_bf16_kernel<false><<<g_ff1, T, 0, stream>>>(hb, w1b, ff_b1, nullptr, t1b, DD, FDIM, 1.0f, 2);
  gemm_bf16_kernel<false><<<g_qkv, T, 0, stream>>>(t1b, w2b, ff_b2, tA, nullptr, FDIM, DD, 1.0f, 0);
  add_ln_kernel<<<NN, T, 0, stream>>>(tB, tA, ln4_g, ln4_b, o_recon, hb);

  // ---- head + gather ----
  gemm_bf16_kernel<false><<<g_qkv, T, 0, stream>>>(hb, hwb, head_b, tA, nullptr, DD, DD, 1.0f, 0);
  gather_out_kernel<<<(MM * KIDX) / T, T, 0, stream>>>(tA, mask_nd, gene_idx, o_xrec);
}
